// ProgressNL_6940667150706
// MI455X (gfx1250) — matrix operand order repaired and checked by reading, not compiled, not run on hardware
//
#include <hip/hip_runtime.h>

namespace {

constexpr int kN = 2, kC = 1024, kTHW = 6272, kDi = 512, kQkvRows = 1536;
constexpr int kIters = kTHW / 32;  // 196 key blocks of 32
constexpr float kEps = 1e-5f;
constexpr float kScale = 0.044194173824159216f;  // 1/sqrt(512)

typedef __bf16 bf16;
typedef bf16  v8bf  __attribute__((ext_vector_type(8)));
typedef bf16  v16bf __attribute__((ext_vector_type(16)));
typedef float v8f   __attribute__((ext_vector_type(8)));
typedef unsigned int       u32x4 __attribute__((ext_vector_type(4)));
typedef int                i32x4 __attribute__((ext_vector_type(4)));
typedef int                i32x8 __attribute__((ext_vector_type(8)));
typedef unsigned long long u64x4 __attribute__((ext_vector_type(4)));

__device__ __forceinline__ v16bf cat16(v8bf lo, v8bf hi) {
  return __builtin_shufflevector(lo, hi, 0,1,2,3,4,5,6,7,8,9,10,11,12,13,14,15);
}
// A-matrix 16x32 bf16 fragment (ISA 7.12.2 layout)
__device__ __forceinline__ v16bf load_a_frag(const bf16* base, int rstride, int lane) {
  const bf16* p = base + (size_t)(lane & 15) * rstride + ((lane >> 4) << 3);
  return cat16(*(const v8bf*)p, *(const v8bf*)(p + 16));
}
// B-matrix 32x16 bf16 fragment: lane holds column (lane&15), K at 8*(lane>>4)+[0,8) and 16+8*(lane>>4)+[0,8)
__device__ __forceinline__ v16bf load_b_frag(const bf16* base, size_t cstride, int lane) {
  const bf16* p = base + (size_t)(lane & 15) * cstride + ((lane >> 4) << 3);
  return cat16(*(const v8bf*)p, *(const v8bf*)(p + 16));
}
__device__ __forceinline__ v8f wmma_bf16(v16bf a, v16bf b, v8f c) {
  return __builtin_amdgcn_wmma_f32_16x16x32_bf16(false, a, false, b, (short)0, c, false, false);
}

// ---------------- TDM descriptor helpers (ISA ch.8 D# layout) ----------------
#if __has_builtin(__builtin_amdgcn_tensor_load_to_lds) && \
    __has_builtin(__builtin_amdgcn_s_wait_tensorcnt)
#define USE_TDM 1
#else
#define USE_TDM 0
#endif

#if USE_TDM
__device__ __forceinline__ u32x4 tdm_g0(unsigned lds_addr, const void* gp) {
  unsigned long long ga = (unsigned long long)(uintptr_t)gp;
  u32x4 g;
  g.x = 1u;  // count=1, user descriptor
  g.y = lds_addr;
  g.z = (unsigned)(ga & 0xFFFFFFFFu);                      // global_addr[31:0]
  g.w = (unsigned)((ga >> 32) & 0x01FFFFFFu) | (2u << 30); // global_addr[56:32] | type=2
  return g;
}
// data_size=3 (8-byte units). tile1==0 -> 1-D copy.
__device__ __forceinline__ i32x8 tdm_g1(unsigned tdim0, unsigned tdim1,
                                        unsigned tile0, unsigned tile1,
                                        unsigned long long stride0) {
  unsigned long long q0 = (3ull << 16) | ((unsigned long long)(tdim0 & 0xFFFFu) << 48);
  unsigned long long q1 = (unsigned long long)((tdim0 >> 16) & 0xFFFFu) |
                          ((unsigned long long)tdim1 << 16) |
                          ((unsigned long long)(tile0 & 0xFFFFu) << 48);
  unsigned long long q2 = (unsigned long long)(tile1 & 0xFFFFu) |
                          ((stride0 & 0xFFFFFFFFull) << 32);
  unsigned long long q3 = (stride0 >> 32) & 0xFFFFull;
  u64x4 q = {q0, q1, q2, q3};
  return __builtin_bit_cast(i32x8, q);
}
__device__ __forceinline__ void tdm_issue(u32x4 g0, i32x8 g1) {
  i32x4 z4 = {0, 0, 0, 0};
#if defined(__clang_major__) && (__clang_major__ >= 23)
  i32x8 z8 = {0, 0, 0, 0, 0, 0, 0, 0};
  __builtin_amdgcn_tensor_load_to_lds(g0, g1, z4, z4, z8, 0);
#else
  __builtin_amdgcn_tensor_load_to_lds(g0, g1, z4, z4, 0);
#endif
}
// 1-D contiguous copy of `units` 8-byte chunks
__device__ __forceinline__ void tdm_load_1d(unsigned lds_off, const void* gp, unsigned units) {
  tdm_issue(tdm_g0(lds_off, gp), tdm_g1(units, 0u, units, 0u, (unsigned long long)units));
}
// 2-D tile: tile1 rows of tile0 8-byte units, row stride stride0 units; packed to LDS
__device__ __forceinline__ void tdm_load_2d(unsigned lds_off, const void* gp, unsigned tile0,
                                            unsigned tile1, unsigned long long stride0) {
  tdm_issue(tdm_g0(lds_off, gp), tdm_g1(tile0, tile1, tile0, tile1, stride0));
}
#endif  // USE_TDM

// ---------------- prep: weights/bias -> bf16 ----------------
__global__ __launch_bounds__(256) void k_wprep(
    const float* __restrict__ wt, const float* __restrict__ wp, const float* __restrict__ wg,
    const float* __restrict__ bt, const float* __restrict__ bp, const float* __restrict__ bg,
    const float* __restrict__ wo,
    bf16* __restrict__ Wqkv, bf16* __restrict__ WoutBf, float* __restrict__ bqkv) {
  int i0 = blockIdx.x * blockDim.x + threadIdx.x;
  int stride = gridDim.x * blockDim.x;
  for (int i = i0; i < kQkvRows * kC; i += stride) {
    int row = i >> 10;
    float v = (row < 512) ? wt[i] : (row < 1024 ? wp[i - 512 * kC] : wg[i - 1024 * kC]);
    Wqkv[i] = (bf16)v;
  }
  for (int i = i0; i < kC * kDi; i += stride) WoutBf[i] = (bf16)wo[i];
  for (int i = i0; i < kQkvRows; i += stride)
    bqkv[i] = (i < 512) ? bt[i] : (i < 1024 ? bp[i - 512] : bg[i - 1024]);
}

// ---------------- prep: x [n][c][t] f32 -> xbT [n][t][c] bf16 ----------------
__global__ __launch_bounds__(256) void k_xpose(const float* __restrict__ x,
                                               bf16* __restrict__ xbT) {
  __shared__ float tile[32][33];
  int n = blockIdx.z;
  int t0 = blockIdx.x * 32, c0 = blockIdx.y * 32;
  int tx = threadIdx.x, ty = threadIdx.y;
#pragma unroll
  for (int i = 0; i < 4; ++i)
    tile[ty + 8 * i][tx] = x[((size_t)n * kC + c0 + ty + 8 * i) * kTHW + t0 + tx];
  __syncthreads();
#pragma unroll
  for (int i = 0; i < 4; ++i)
    xbT[((size_t)n * kTHW + t0 + ty + 8 * i) * kC + c0 + tx] = (bf16)tile[tx][ty + 8 * i];
}

// ---------------- QKV projection GEMM ----------------
__global__ __launch_bounds__(256) void k_qkv(const bf16* __restrict__ Wqkv,
                                             const bf16* __restrict__ xbT,
                                             const float* __restrict__ bqkv,
                                             bf16* __restrict__ thetaT,
                                             bf16* __restrict__ phiT,
                                             bf16* __restrict__ gD) {
  int n = blockIdx.z;
  int t0 = blockIdx.x * 16;
  int lane = threadIdx.x & 31, wave = threadIdx.x >> 5;
  int d0 = (blockIdx.y * 8 + wave) * 16;

  const bf16* Abase = Wqkv + (size_t)d0 * kC;
  const bf16* Bbase = xbT + ((size_t)n * kTHW + t0) * kC;

  v8f acc = {0.f, 0.f, 0.f, 0.f, 0.f, 0.f, 0.f, 0.f};
#pragma unroll 4
  for (int k0 = 0; k0 < kC; k0 += 32) {
    v16bf a = load_a_frag(Abase + k0, kC, lane);
    v16bf b = load_b_frag(Bbase + k0, kC, lane);
    acc = wmma_bf16(a, b, acc);
  }
  int t = t0 + (lane & 15);
  int rbase = d0 + 8 * (lane >> 4);
#pragma unroll
  for (int j = 0; j < 8; ++j) {
    int d = rbase + j;
    float v = acc[j] + bqkv[d];
    if (d < 512)
      thetaT[((size_t)n * kTHW + t) * kDi + d] = (bf16)v;
    else if (d < 1024)
      phiT[((size_t)n * kTHW + t) * kDi + (d - 512)] = (bf16)v;
    else
      gD[((size_t)n * kDi + (d - 1024)) * kTHW + t] = (bf16)v;
  }
}

// ---------------- fused flash attention with TDM streaming ----------------
// LDS pool (dynamic, 160 KB total, assumed at LDS offset 0):
//   [0      ,  32768) phi buf 0   [32 tokens][512 d] bf16
//   [32768  ,  65536) phi buf 1
//   [65536  ,  98304) g   buf 0   [512 d][32 tokens] bf16
//   [98304  , 131072) g   buf 1
//   [131072 , 147456) theta tile  [16][512] bf16
//   [147456 , 163840) slab: 8 x [16x32] f32 partial S; S_red/P/alpha carved out
constexpr unsigned kOffPhi0 = 0;
constexpr unsigned kOffG0 = 65536;
constexpr unsigned kOffTheta = 131072, kOffSlab = 147456;
constexpr unsigned kAttnLds = 163840;

__global__ __launch_bounds__(256) void k_attn(const bf16* __restrict__ thetaT,
                                              const bf16* __restrict__ phiT,
                                              const bf16* __restrict__ gD,
                                              bf16* __restrict__ oT) {
  extern __shared__ char smem[];
  bf16* theta_s = (bf16*)(smem + kOffTheta);
  float* slab = (float*)(smem + kOffSlab);          // 8*512 floats
  float* S_red = slab;                              // reuse: written after slab reads
  bf16* P_s = (bf16*)(smem + kOffSlab + 2048);      // 512 bf16
  float* alpha_s = (float*)(smem + kOffSlab + 3072);// 16 floats

  int n = blockIdx.y, t0 = blockIdx.x * 16;
  int tid = threadIdx.x, lane = tid & 31, wave = tid >> 5;
  int dbase = wave * 64;

  const bf16* theta_g = thetaT + ((size_t)n * kTHW + t0) * kDi;
  const bf16* phi_g = phiT + (size_t)n * kTHW * kDi;
  const bf16* g_g = gD + (size_t)n * kDi * kTHW;

#if USE_TDM
  if (wave == 0) {
    tdm_load_1d(kOffTheta, theta_g, 16 * kDi * 2 / 8);         // 2048 units
    tdm_load_1d(kOffPhi0, phi_g, 32 * kDi * 2 / 8);            // 4096 units
    tdm_load_2d(kOffG0, g_g, 8, 512, (unsigned long long)(kTHW * 2 / 8));
  }
#else
  for (int i = tid; i < 16 * kDi / 8; i += 256)
    ((v8bf*)theta_s)[i] = ((const v8bf*)theta_g)[i];
#endif

  float m_run = -1e30f, l_run = 0.f;  // live in softmax threads (tid<16)
  v8f o_acc[4];
#pragma unroll
  for (int s = 0; s < 4; ++s) o_acc[s] = (v8f){0.f, 0.f, 0.f, 0.f, 0.f, 0.f, 0.f, 0.f};

  for (int it = 0; it < kIters; ++it) {
    unsigned buf = (unsigned)(it & 1);
    bf16* phi_b = (bf16*)(smem + kOffPhi0 + buf * 32768u);
    bf16* g_b = (bf16*)(smem + kOffG0 + buf * 32768u);
#if USE_TDM
    if (wave == 0) {
      if (it + 1 < kIters) {
        int p1 = (it + 1) * 32;
        unsigned nb = (unsigned)((it + 1) & 1);
        tdm_load_1d(kOffPhi0 + nb * 32768u, phi_g + (size_t)p1 * kDi, 4096u);
        tdm_load_2d(kOffG0 + nb * 32768u, g_g + p1, 8u, 512u, 1568ull);
        __builtin_amdgcn_s_wait_tensorcnt(2);  // current pair (and prologue) done
      } else {
        __builtin_amdgcn_s_wait_tensorcnt(0);
      }
    }
#else
    {
      int p0 = it * 32;
      for (int i = tid; i < 32 * kDi / 8; i += 256)
        ((v8bf*)phi_b)[i] = ((const v8bf*)(phi_g + (size_t)p0 * kDi))[i];
      for (int i = tid; i < 512 * 32 / 4; i += 256) {
        int d = i >> 3, part = i & 7;
        ((unsigned long long*)g_b)[i] =
            ((const unsigned long long*)(g_g + (size_t)d * kTHW + p0))[part];
      }
    }
#endif
    __syncthreads();  // tiles for this iteration visible

    // ---- partial scores: this wave's 64-dim K slice ----
    v8f sacc[2];
    sacc[0] = (v8f){0.f, 0.f, 0.f, 0.f, 0.f, 0.f, 0.f, 0.f};
    sacc[1] = sacc[0];
#pragma unroll
    for (int ks = 0; ks < 2; ++ks) {
      v16bf a = load_a_frag(theta_s + dbase + 32 * ks, kDi, lane);
#pragma unroll
      for (int ns = 0; ns < 2; ++ns) {
        v16bf b = load_b_frag(phi_b + (size_t)(16 * ns) * kDi + dbase + 32 * ks,
                              kDi, lane);
        sacc[ns] = wmma_bf16(a, b, sacc[ns]);
      }
    }
    {
      int m = 8 * (lane >> 4), nn = lane & 15;
#pragma unroll
      for (int ns = 0; ns < 2; ++ns)
#pragma unroll
        for (int j = 0; j < 8; ++j)
          slab[wave * 512 + (m + j) * 32 + 16 * ns + nn] = sacc[ns][j];
    }
    __syncthreads();

    // ---- cross-wave K reduction (reads all slabs, then writes S_red=slab[0]) ----
    for (int e = tid; e < 512; e += 256) {
      float s = 0.f;
#pragma unroll
      for (int w = 0; w < 8; ++w) s += slab[w * 512 + e];
      S_red[e] = s * kScale;
    }
    __syncthreads();

    // ---- online softmax (one thread per query row) ----
    if (tid < 16) {
      int r = tid;
      float mloc = -1e30f;
#pragma unroll 4
      for (int q = 0; q < 32; ++q) mloc = fmaxf(mloc, S_red[r * 32 + q]);
      float m_new = fmaxf(m_run, mloc);
      float alpha = __expf(m_run - m_new);
      float ls = 0.f;
#pragma unroll 4
      for (int q = 0; q < 32; ++q) {
        float e = __expf(S_red[r * 32 + q] - m_new);
        ls += e;
        P_s[r * 32 + q] = (bf16)e;
      }
      l_run = l_run * alpha + ls;
      m_run = m_new;
      alpha_s[r] = alpha;
    }
    __syncthreads();

    // ---- O rescale + O += P x g^T (this wave's 64 dims) ----
    {
      float al[8];
#pragma unroll
      for (int j = 0; j < 8; ++j) al[j] = alpha_s[j + 8 * (lane >> 4)];
      v16bf pa = load_a_frag(P_s, 32, lane);
#pragma unroll
      for (int ns = 0; ns < 4; ++ns) {
#pragma unroll
        for (int j = 0; j < 8; ++j) o_acc[ns][j] *= al[j];
        v16bf b = load_b_frag(g_b + (size_t)(dbase + 16 * ns) * 32, 32, lane);
        o_acc[ns] = wmma_bf16(pa, b, o_acc[ns]);
      }
    }
    __syncthreads();  // buf consumed; next iter may DMA over the other buffer
  }

  // ---- epilogue: divide by row sums, store oT [n][t][d] bf16 ----
  if (tid < 16) S_red[tid] = 1.0f / l_run;
  __syncthreads();
  float inv[8];
#pragma unroll
  for (int j = 0; j < 8; ++j) inv[j] = S_red[j + 8 * (lane >> 4)];
#pragma unroll
  for (int ns = 0; ns < 4; ++ns) {
    int d = dbase + 16 * ns + (lane & 15);
#pragma unroll
    for (int j = 0; j < 8; ++j) {
      int t = t0 + j + 8 * (lane >> 4);
      oT[((size_t)n * kTHW + t) * kDi + d] = (bf16)(o_acc[ns][j] * inv[j]);
    }
  }
}

// ---------------- output projection: p[n][c][t] = w_out @ o + b_out ----------------
__global__ __launch_bounds__(256) void k_outgemm(const bf16* __restrict__ oT,
                                                 const bf16* __restrict__ WoutBf,
                                                 const float* __restrict__ bout,
                                                 float* __restrict__ p) {
  int n = blockIdx.z;
  int t0 = blockIdx.x * 16;
  int lane = threadIdx.x & 31, wave = threadIdx.x >> 5;
  int c0 = (blockIdx.y * 8 + wave) * 16;

  const bf16* Abase = oT + ((size_t)n * kTHW + t0) * kDi;
  const bf16* Bbase = WoutBf + (size_t)c0 * kDi;

  v8f acc = {0.f, 0.f, 0.f, 0.f, 0.f, 0.f, 0.f, 0.f};
#pragma unroll 4
  for (int k0 = 0; k0 < kDi; k0 += 32) {
    v16bf a = load_a_frag(Abase + k0, kDi, lane);
    v16bf b = load_b_frag(Bbase + k0, kDi, lane);
    acc = wmma_bf16(a, b, acc);
  }
  int c = c0 + (lane & 15);
  float bias = bout[c];
#pragma unroll
  for (int j = 0; j < 8; ++j) {
    int t = t0 + j + 8 * (lane >> 4);
    p[((size_t)n * kC + c) * kTHW + t] = acc[j] + bias;
  }
}

// ---------------- deterministic mean/var reduction ----------------
__global__ __launch_bounds__(256) void k_stats(const float* __restrict__ p,
                                               float* __restrict__ part) {
  __shared__ float s1[256], s2[256];
  int n = blockIdx.y, tid = threadIdx.x;
  const float* base = p + (size_t)n * kC * kTHW;
  float a = 0.f, b = 0.f;
  for (size_t i = (size_t)blockIdx.x * 256 + tid; i < (size_t)kC * kTHW; i += (size_t)64 * 256) {
    float v = base[i];
    a += v;
    b += v * v;
  }
  s1[tid] = a; s2[tid] = b;
  __syncthreads();
  for (int s = 128; s > 0; s >>= 1) {
    if (tid < s) { s1[tid] += s1[tid + s]; s2[tid] += s2[tid + s]; }
    __syncthreads();
  }
  if (tid == 0) {
    part[(n * 64 + blockIdx.x) * 2 + 0] = s1[0];
    part[(n * 64 + blockIdx.x) * 2 + 1] = s2[0];
  }
}

__global__ __launch_bounds__(64) void k_stats2(const float* __restrict__ part,
                                               float* __restrict__ stats) {
  __shared__ float s1[64], s2[64];
  int n = blockIdx.x, tid = threadIdx.x;
  s1[tid] = part[(n * 64 + tid) * 2 + 0];
  s2[tid] = part[(n * 64 + tid) * 2 + 1];
  __syncthreads();
  for (int s = 32; s > 0; s >>= 1) {
    if (tid < s) { s1[tid] += s1[tid + s]; s2[tid] += s2[tid + s]; }
    __syncthreads();
  }
  if (tid == 0) { stats[2 * n] = s1[0]; stats[2 * n + 1] = s2[0]; }
}

// ---------------- groupnorm + residual ----------------
__global__ __launch_bounds__(256) void k_final(const float* __restrict__ x,
                                               const float* __restrict__ p,
                                               const float* __restrict__ stats,
                                               float* __restrict__ out) {
  constexpr float invM = 1.0f / ((float)kC * (float)kTHW);
  int total = kN * kC * kTHW;
  for (int i = blockIdx.x * blockDim.x + threadIdx.x; i < total;
       i += gridDim.x * blockDim.x) {
    int n = i / (kC * kTHW);
    float mean = stats[2 * n] * invM;
    float var = stats[2 * n + 1] * invM - mean * mean;
    out[i] = x[i] + (p[i] - mean) * rsqrtf(var + kEps);
  }
}

// ---------------- workspace layout ----------------
constexpr size_t alg(size_t v) { return (v + 255) & ~(size_t)255; }
constexpr size_t SZ_XBT  = (size_t)kN * kTHW * kC * 2;
constexpr size_t SZ_WQKV = (size_t)kQkvRows * kC * 2;
constexpr size_t SZ_WOUT = (size_t)kC * kDi * 2;
constexpr size_t SZ_BQKV = (size_t)kQkvRows * 4;
constexpr size_t SZ_TOK  = (size_t)kN * kTHW * kDi * 2;
constexpr size_t SZ_P    = (size_t)kN * kC * kTHW * 4;
constexpr size_t SZ_PART = (size_t)kN * 64 * 2 * 4;

constexpr size_t OFF_XBT   = 0;
constexpr size_t OFF_WQKV  = OFF_XBT + alg(SZ_XBT);
constexpr size_t OFF_WOUT  = OFF_WQKV + alg(SZ_WQKV);
constexpr size_t OFF_BQKV  = OFF_WOUT + alg(SZ_WOUT);
constexpr size_t OFF_THETA = OFF_BQKV + alg(SZ_BQKV);
constexpr size_t OFF_PHI   = OFF_THETA + alg(SZ_TOK);
constexpr size_t OFF_G     = OFF_PHI + alg(SZ_TOK);
constexpr size_t OFF_OT    = OFF_G + alg(SZ_TOK);
constexpr size_t OFF_PBUF  = OFF_OT + alg(SZ_TOK);
constexpr size_t OFF_PART  = OFF_PBUF + alg(SZ_P);
constexpr size_t OFF_STAT  = OFF_PART + alg(SZ_PART);

}  // namespace

extern "C" void kernel_launch(void* const* d_in, const int* in_sizes, int n_in,
                              void* d_out, int out_size, void* d_ws, size_t ws_size,
                              hipStream_t stream) {
  const float* x  = (const float*)d_in[0];
  const float* wt = (const float*)d_in[1];
  const float* bt = (const float*)d_in[2];
  const float* wp = (const float*)d_in[3];
  const float* bp = (const float*)d_in[4];
  const float* wg = (const float*)d_in[5];
  const float* bg = (const float*)d_in[6];
  const float* wo = (const float*)d_in[7];
  const float* bo = (const float*)d_in[8];
  float* out = (float*)d_out;

  char* ws = (char*)d_ws;
  bf16*  xbT    = (bf16*)(ws + OFF_XBT);
  bf16*  Wqkv   = (bf16*)(ws + OFF_WQKV);
  bf16*  WoutBf = (bf16*)(ws + OFF_WOUT);
  float* bqkv   = (float*)(ws + OFF_BQKV);
  bf16*  thetaT = (bf16*)(ws + OFF_THETA);
  bf16*  phiT   = (bf16*)(ws + OFF_PHI);
  bf16*  gD     = (bf16*)(ws + OFF_G);
  bf16*  oT     = (bf16*)(ws + OFF_OT);
  float* pbuf   = (float*)(ws + OFF_PBUF);
  float* part   = (float*)(ws + OFF_PART);
  float* stats  = (float*)(ws + OFF_STAT);

  k_wprep<<<dim3(512), dim3(256), 0, stream>>>(wt, wp, wg, bt, bp, bg, wo,
                                               Wqkv, WoutBf, bqkv);
  k_xpose<<<dim3(kTHW / 32, kC / 32, kN), dim3(32, 8), 0, stream>>>(x, xbT);
  k_qkv<<<dim3(kTHW / 16, kQkvRows / 128, kN), dim3(256), 0, stream>>>(
      Wqkv, xbT, bqkv, thetaT, phiT, gD);
  k_attn<<<dim3(kTHW / 16, kN), dim3(256), kAttnLds, stream>>>(thetaT, phiT, gD, oT);
  k_outgemm<<<dim3(kTHW / 16, kC / 128, kN), dim3(256), 0, stream>>>(
      oT, WoutBf, bo, pbuf);
  k_stats<<<dim3(64, kN), dim3(256), 0, stream>>>(pbuf, part);
  k_stats2<<<dim3(kN), dim3(64), 0, stream>>>(part, stats);
  k_final<<<dim3(2048), dim3(256), 0, stream>>>(x, pbuf, stats, out);
}
